// MultiHeadSelfAttentionBlock_25692494365164
// MI455X (gfx1250) — matrix operand order repaired and checked by reading, not compiled, not run on hardware
//
#include <hip/hip_runtime.h>
#include <hip/hip_bf16.h>

// ---------------------------------------------------------------------------
// MultiHeadSelfAttentionBlock for MI455X (gfx1250, wave32, WMMA + async LDS)
//   B=8, N=1024, E=768, H=12, HD=64
// LN -> QKV GEMM (bf16 wmma, async-LDS B staging, bf16 out)
//    -> flash attention (bf16 wmma, async-LDS K staging, DPP reductions)
//    -> out-proj GEMM (bf16 wmma, f32 out)
// ---------------------------------------------------------------------------

typedef __bf16 bf16_t;
typedef __attribute__((ext_vector_type(16))) __bf16 v16bf;
typedef __attribute__((ext_vector_type(8)))  __bf16 v8bf;
typedef __attribute__((ext_vector_type(8)))  float  v8f;

union AB16 { v16bf v; v8bf h[2]; __bf16 e[16]; };

__device__ __forceinline__ v8f wmma_bf16(v16bf a, v16bf b, v8f c) {
  // D = A(16x32 bf16) * B(32x16 bf16) + C(16x16 f32)
  return __builtin_amdgcn_wmma_f32_16x16x32_bf16(
      false, a, false, b, (short)0, c, false, false);
}

// async global->LDS copy of 16 bytes (per lane); tracked by ASYNCcnt
__device__ __forceinline__ void async_copy_b128(uint32_t lds_addr,
                                                const void* gptr) {
  asm volatile("global_load_async_to_lds_b128 %0, %1, off"
               :: "v"(lds_addr), "v"(gptr) : "memory");
}
__device__ __forceinline__ void wait_async0() {
  asm volatile("s_wait_asynccnt 0x0" ::: "memory");
}
__device__ __forceinline__ uint32_t lds_off(const void* p) {
  return (uint32_t)(uintptr_t)p;  // flat-LDS addr[31:0] == LDS offset
}

// ---- DPP 16-lane (one DPP row) reductions: pure VALU, no LDS traffic ------
#define DPP_QUAD(a, b, c, d) ((a) | ((b) << 2) | ((c) << 4) | ((d) << 6))
#define DPP_ROW_MIRROR      0x140
#define DPP_ROW_HALF_MIRROR 0x141

template <int CTRL>
__device__ __forceinline__ float dpp_mov(float x) {
  return __builtin_bit_cast(
      float, __builtin_amdgcn_update_dpp(0, __builtin_bit_cast(int, x), CTRL,
                                         0xF, 0xF, true));
}
__device__ __forceinline__ float redmax16(float x) {
  x = fmaxf(x, dpp_mov<DPP_ROW_MIRROR>(x));       // pairs (i, 15-i)
  x = fmaxf(x, dpp_mov<DPP_ROW_HALF_MIRROR>(x));  // pairs (i, 7-i) per 8
  x = fmaxf(x, dpp_mov<DPP_QUAD(2, 3, 0, 1)>(x)); // xor 2
  x = fmaxf(x, dpp_mov<DPP_QUAD(1, 0, 3, 2)>(x)); // xor 1
  return x;
}
__device__ __forceinline__ float redsum16(float x) {
  x += dpp_mov<DPP_ROW_MIRROR>(x);
  x += dpp_mov<DPP_ROW_HALF_MIRROR>(x);
  x += dpp_mov<DPP_QUAD(2, 3, 0, 1)>(x);
  x += dpp_mov<DPP_QUAD(1, 0, 3, 2)>(x);
  return x;
}

// ---------------------------------------------------------------------------
// f32 -> bf16 elementwise convert (weights)
// ---------------------------------------------------------------------------
__global__ void f32_to_bf16_kernel(const float* __restrict__ in,
                                   bf16_t* __restrict__ out, int n) {
  int i = blockIdx.x * blockDim.x + threadIdx.x;
  if (i < n) out[i] = (bf16_t)in[i];
}

// ---------------------------------------------------------------------------
// LayerNorm over E=768, one block (256 threads) per row; bf16 output
// ---------------------------------------------------------------------------
__global__ __launch_bounds__(256)
void layernorm_bf16_kernel(const float* __restrict__ x,
                           const float* __restrict__ g,
                           const float* __restrict__ b,
                           bf16_t* __restrict__ out) {
  const int row = blockIdx.x;
  const int tid = threadIdx.x;
  const float* xr = x + (size_t)row * 768;
  float v0 = xr[tid], v1 = xr[tid + 256], v2 = xr[tid + 512];
  float s  = v0 + v1 + v2;
  float s2 = v0 * v0 + v1 * v1 + v2 * v2;
#pragma unroll
  for (int m = 16; m; m >>= 1) {
    s  += __shfl_xor(s,  m, 32);
    s2 += __shfl_xor(s2, m, 32);
  }
  __shared__ float ps[8], ps2[8];
  const int wave = tid >> 5, lane = tid & 31;
  if (lane == 0) { ps[wave] = s; ps2[wave] = s2; }
  __syncthreads();
  if (wave == 0) {
    float a1 = (lane < 8) ? ps[lane]  : 0.f;
    float a2 = (lane < 8) ? ps2[lane] : 0.f;
#pragma unroll
    for (int m = 4; m; m >>= 1) {
      a1 += __shfl_xor(a1, m, 32);
      a2 += __shfl_xor(a2, m, 32);
    }
    if (lane == 0) { ps[0] = a1; ps2[0] = a2; }
  }
  __syncthreads();
  const float mean = ps[0] * (1.f / 768.f);
  const float var  = ps2[0] * (1.f / 768.f) - mean * mean;
  const float inv  = rsqrtf(var + 1e-5f);
  bf16_t* orow = out + (size_t)row * 768;
  orow[tid]       = (bf16_t)((v0 - mean) * inv * g[tid]       + b[tid]);
  orow[tid + 256] = (bf16_t)((v1 - mean) * inv * g[tid + 256] + b[tid + 256]);
  orow[tid + 512] = (bf16_t)((v2 - mean) * inv * g[tid + 512] + b[tid + 512]);
}

// ---------------------------------------------------------------------------
// C[M,N] = A[M,K](bf16) * W[N,K]^T(bf16) + bias[N]  (OutT = f32 or bf16)
// 8 waves/block, 128x64 block tile, K step 32.
// B tile double-buffered in LDS via async global->LDS prefetch.
// ---------------------------------------------------------------------------
template <typename OutT>
__global__ __launch_bounds__(256)
void gemm_bf16_wmma_kernel(const bf16_t* __restrict__ A,
                           const bf16_t* __restrict__ W,
                           const float* __restrict__ bias,
                           OutT* __restrict__ C,
                           int M, int N, int K) {
  __shared__ alignas(32) bf16_t Bs[2][64 * 32];  // [buf][n-in-tile][k]
  const int tid  = threadIdx.x;
  const int wave = tid >> 5;
  const int lane = tid & 31;
  const int half = lane >> 4;
  const int ln16 = lane & 15;

  const int rowBase = blockIdx.x * 128 + wave * 16;
  const int colBase = blockIdx.y * 64;

  v8f acc[4] = {v8f{}, v8f{}, v8f{}, v8f{}};

  const bf16_t* aPtr = A + (size_t)(rowBase + ln16) * K;

  // cooperative B staging: 64 n x 32 k, one 16B async chunk per thread
  const int sn = tid >> 2;        // 0..63
  const int sk = (tid & 3) * 8;   // 0,8,16,24
  const bf16_t* wPtr = W + (size_t)(colBase + sn) * K + sk;
  const uint32_t bDst[2] = {lds_off(&Bs[0][sn * 32 + sk]),
                            lds_off(&Bs[1][sn * 32 + sk])};

  async_copy_b128(bDst[0], wPtr);  // prefetch first tile

  int buf = 0;
  for (int k0 = 0; k0 < K; k0 += 32, buf ^= 1) {
    wait_async0();        // own chunk of current tile landed
    __syncthreads();      // everyone's chunks landed; prev reads done
    if (k0 + 32 < K)      // prefetch next tile into other buffer
      async_copy_b128(bDst[buf ^ 1], wPtr + k0 + 32);

    AB16 a;  // A frag: K = 16*half+[0..16)
    a.h[0] = *(const v8bf*)(aPtr + k0 + 16 * half);
    a.h[1] = *(const v8bf*)(aPtr + k0 + 16 * half + 8);

    const bf16_t* bs = &Bs[buf][0];
#pragma unroll
    for (int j = 0; j < 4; ++j) {
      v16bf bfr = *(const v16bf*)&bs[(j * 16 + ln16) * 32 + 16 * half];
      acc[j] = wmma_bf16(a.v, bfr, acc[j]);
    }
  }

#pragma unroll
  for (int j = 0; j < 4; ++j) {
    const int col = colBase + j * 16 + ln16;
    const float bv = bias[col];
#pragma unroll
    for (int v = 0; v < 8; ++v) {
      const int row = rowBase + v + 8 * half;
      C[(size_t)row * N + col] = (OutT)(acc[j][v] + bv);
    }
  }
}

// ---------------------------------------------------------------------------
// Flash attention over qkv[8192, 2304] (bf16).
// grid: (N/128, B*H); 8 waves/block, wave owns 16 q rows.
// K tile staged via async global->LDS; V^T staged bf16 (transpose);
// softmax row reductions via DPP (no LDS); P staged bf16.
// ---------------------------------------------------------------------------
#define KS_STRIDE 80  // bf16 elems; 160B pitch: 32B-aligned rows, bank spread
#define VT_STRIDE 80
#define PS_STRIDE 72  // bf16 elems; 144B pitch keeps every v8bf read 16B-aligned

__global__ __launch_bounds__(256)
void attention_wmma_kernel(const bf16_t* __restrict__ qkv,
                           bf16_t* __restrict__ ctx) {
  const int bh = blockIdx.y;
  const int b = bh / 12, h = bh % 12;
  const int tid  = threadIdx.x;
  const int wave = tid >> 5;
  const int lane = tid & 31;
  const int half = lane >> 4;
  const int ln16 = lane & 15;

  __shared__ alignas(32) bf16_t Ks[64 * KS_STRIDE];    // K tile: [key][hd]
  __shared__ alignas(32) bf16_t Vt[64 * VT_STRIDE];    // V^T:    [hd][key]
  __shared__ alignas(32) bf16_t Ps[8][16 * PS_STRIDE]; // per-wave bf16 P

  // Q fragments (scale 1/8 folded in; exact in bf16), kept in registers
  const int qRow = blockIdx.x * 128 + wave * 16 + ln16;
  const bf16_t* qPtr = qkv + (size_t)(b * 1024 + qRow) * 2304 + h * 64;
  AB16 aq[2];
#pragma unroll
  for (int c = 0; c < 2; ++c) {
#pragma unroll
    for (int i = 0; i < 8; ++i) {
      aq[c].e[i]     = (bf16_t)((float)qPtr[c * 32 + 16 * half + i] * 0.125f);
      aq[c].e[8 + i] = (bf16_t)((float)qPtr[c * 32 + 16 * half + 8 + i] * 0.125f);
    }
  }

  v8f o[4] = {v8f{}, v8f{}, v8f{}, v8f{}};
  float mrow[8], lrow[8];
#pragma unroll
  for (int v = 0; v < 8; ++v) { mrow[v] = -1e30f; lrow[v] = 0.f; }

  const bf16_t* kvBase = qkv + (size_t)(b * 1024) * 2304;

  // K staging: thread -> one key row, 16 hd values (two async b128)
  const int sKey = tid >> 2;          // 0..63
  const int sHd  = (tid & 3) * 16;    // 0,16,32,48
  const bf16_t* kSrc = kvBase + (size_t)sKey * 2304 + 768 + h * 64 + sHd;
  const uint32_t kDst = lds_off(&Ks[sKey * KS_STRIDE + sHd]);

  for (int kc = 0; kc < 1024; kc += 64) {
    __syncthreads();  // all reads of Ks/Vt from previous iter done
    {   // async K tile (pure byte copy, B-fragment friendly layout)
      const bf16_t* ks = kSrc + (size_t)kc * 2304;
      async_copy_b128(kDst,      ks);
      async_copy_b128(kDst + 16, ks + 8);
    }
    {   // V^T staging (bf16 copies, transposed)
      const int key = tid & 63;
      const int hdB = (tid >> 6) * 16;
      const bf16_t* vp = kvBase + (size_t)(kc + key) * 2304 + 1536 + h * 64 + hdB;
      AB16 vv;
      vv.h[0] = *(const v8bf*)vp;
      vv.h[1] = *(const v8bf*)(vp + 8);
#pragma unroll
      for (int i = 0; i < 16; ++i)
        Vt[(hdB + i) * VT_STRIDE + key] = vv.e[i];
    }
    wait_async0();
    __syncthreads();

    // S = Q * K^T  (4 tiles of 16 keys)
    v8f s[4];
#pragma unroll
    for (int j = 0; j < 4; ++j) {
      v8f a = {};
#pragma unroll
      for (int c = 0; c < 2; ++c) {
        v16bf bk = *(const v16bf*)&Ks[(j * 16 + ln16) * KS_STRIDE +
                                      c * 32 + 16 * half];
        a = wmma_bf16(aq[c].v, bk, a);
      }
      s[j] = a;
    }

    // online softmax (rows = v + 8*half; DPP reduce across 16-lane row)
    float fac[8];
#pragma unroll
    for (int v = 0; v < 8; ++v) {
      float mx = fmaxf(fmaxf(s[0][v], s[1][v]), fmaxf(s[2][v], s[3][v]));
      mx = redmax16(mx);
      const float mnew = fmaxf(mrow[v], mx);
      const float f = __expf(mrow[v] - mnew);
      float psum = 0.f;
#pragma unroll
      for (int j = 0; j < 4; ++j) {
        const float p = __expf(s[j][v] - mnew);
        s[j][v] = p;
        psum += p;
      }
      psum = redsum16(psum);
      lrow[v] = lrow[v] * f + psum;
      mrow[v] = mnew;
      fac[v]  = f;
    }
#pragma unroll
    for (int j = 0; j < 4; ++j)
#pragma unroll
      for (int v = 0; v < 8; ++v) o[j][v] *= fac[v];

    // C-layout -> A-layout for P via wave-private bf16 LDS staging
    bf16_t* ps = &Ps[wave][0];
#pragma unroll
    for (int j = 0; j < 4; ++j)
#pragma unroll
      for (int v = 0; v < 8; ++v)
        ps[(v + 8 * half) * PS_STRIDE + j * 16 + ln16] = (bf16_t)s[j][v];
    // wave-private region: in-wave DS ordering via dscnt, no barrier needed

    // O += P * V  (A = P[16x32] bf16 from LDS, B = V^T slices from LDS)
#pragma unroll
    for (int c = 0; c < 2; ++c) {
      AB16 ap;
      const bf16_t* pr = ps + ln16 * PS_STRIDE + c * 32;
      ap.h[0] = *(const v8bf*)(pr + 16 * half);       // K = 32c+16*half+[0..8)
      ap.h[1] = *(const v8bf*)(pr + 16 * half + 8);   // K = 32c+16*half+8+[0..8)
#pragma unroll
      for (int j = 0; j < 4; ++j) {
        v16bf bv = *(const v16bf*)&Vt[(j * 16 + ln16) * VT_STRIDE +
                                      c * 32 + 16 * half];
        o[j] = wmma_bf16(ap.v, bv, o[j]);
      }
    }
  }

  // normalize and emit ctx (bf16) at [b*N + q, h*64 + hd]
  const int oRow = b * 1024 + blockIdx.x * 128 + wave * 16;
#pragma unroll
  for (int v = 0; v < 8; ++v) {
    const float rl = 1.f / lrow[v];
#pragma unroll
    for (int j = 0; j < 4; ++j)
      ctx[(size_t)(oRow + v + 8 * half) * 768 + h * 64 + j * 16 + ln16] =
          (bf16_t)(o[j][v] * rl);
  }
}

// ---------------------------------------------------------------------------
extern "C" void kernel_launch(void* const* d_in, const int* in_sizes, int n_in,
                              void* d_out, int out_size, void* d_ws, size_t ws_size,
                              hipStream_t stream) {
  (void)in_sizes; (void)n_in; (void)out_size; (void)ws_size;
  const float* x     = (const float*)d_in[0];
  const float* ln_g  = (const float*)d_in[1];
  const float* ln_b  = (const float*)d_in[2];
  const float* w_qkv = (const float*)d_in[3];  // [2304, 768]
  const float* b_qkv = (const float*)d_in[4];  // [2304]
  const float* w_out = (const float*)d_in[5];  // [768, 768]
  const float* b_out = (const float*)d_in[6];  // [768]
  float* out = (float*)d_out;                  // [8,1024,768]

  char* ws = (char*)d_ws;
  size_t off = 0;
  auto take = [&](size_t bytes) -> void* {
    off = (off + 255) & ~(size_t)255;
    void* p = ws + off;
    off += bytes;
    return p;
  };
  bf16_t* xn     = (bf16_t*)take(8192ull * 768 * 2);   // LN output
  bf16_t* wq_bf  = (bf16_t*)take(2304ull * 768 * 2);   // qkv weights bf16
  bf16_t* wo_bf  = (bf16_t*)take(768ull * 768 * 2);    // out weights bf16
  bf16_t* qkv_bf = (bf16_t*)take(8192ull * 2304 * 2);  // qkv projections bf16
  bf16_t* ctx    = (bf16_t*)take(8192ull * 768 * 2);   // attention ctx bf16

  f32_to_bf16_kernel<<<(2304 * 768 + 255) / 256, 256, 0, stream>>>(w_qkv, wq_bf, 2304 * 768);
  f32_to_bf16_kernel<<<(768 * 768 + 255) / 256, 256, 0, stream>>>(w_out, wo_bf, 768 * 768);
  layernorm_bf16_kernel<<<8192, 256, 0, stream>>>(x, ln_g, ln_b, xn);
  // qkv = xn @ w_qkv^T + b_qkv : M=8192, N=2304, K=768 (bf16 out)
  gemm_bf16_wmma_kernel<bf16_t><<<dim3(64, 36), 256, 0, stream>>>(
      xn, wq_bf, b_qkv, qkv_bf, 8192, 2304, 768);
  // attention per (b,h), 128 q rows per block
  attention_wmma_kernel<<<dim3(8, 96), 256, 0, stream>>>(qkv_bf, ctx);
  // out = ctx @ w_out^T + b_out : M=8192, N=768, K=768 (f32 out)
  gemm_bf16_wmma_kernel<float><<<dim3(64, 12), 256, 0, stream>>>(
      ctx, wo_bf, b_out, out, 8192, 768, 768);
}
